// Attention_5420248728069
// MI455X (gfx1250) — matrix operand order repaired and checked by reading, not compiled, not run on hardware
//
#include <hip/hip_runtime.h>
#include <hip/hip_bf16.h>

// ---------------------------------------------------------------------------
// Fused cross-attention for MI455X (gfx1250, wave32, WMMA bf16 16x16x32).
//   v_value  = v_code  @ Wv^T      obs_value = obs_code @ Wv^T
//   v_query  = v_code  @ Wq^T      v_key     = v_code  @ Wk^T
//   obs_key  = obs_code @ Wk^T
//   S = [q.k_self , v_query @ obs_key^T] / T ; w = softmax(S)
//   y = LN( w0*v_value + w[1:] @ obs_value + v_code )
// Flash-style: never materialize [N, M+1].
// ---------------------------------------------------------------------------

constexpr int kE = 512;
constexpr int kN = 8192;
constexpr int kM = 4096;
constexpr float kInvT = 0.044194173824159216f;  // 1/sqrt(512)
constexpr float kEps  = 1e-6f;

typedef __bf16 bf16_t;
typedef __attribute__((ext_vector_type(16))) __bf16 v16bf;
typedef __attribute__((ext_vector_type(8)))  __bf16 v8bf;
typedef __attribute__((ext_vector_type(8)))  float  v8f;

union BV { v16bf v; v8bf h[2]; };

__device__ __forceinline__ bf16_t f2bf(float f) {
  unsigned u = __builtin_bit_cast(unsigned, f);
  u += 0x7fffu + ((u >> 16) & 1u);                 // round-to-nearest-even
  unsigned short s = (unsigned short)(u >> 16);
  return __builtin_bit_cast(bf16_t, s);
}
__device__ __forceinline__ float bf2f(bf16_t b) {
  unsigned short s = __builtin_bit_cast(unsigned short, b);
  unsigned u = ((unsigned)s) << 16;
  return __builtin_bit_cast(float, u);
}
__device__ __forceinline__ v8f wmma_bf16(v16bf a, v16bf b, v8f c) {
  // (neg_a, A, neg_b, B, c_mod, C, reuse_a, reuse_b)
  return __builtin_amdgcn_wmma_f32_16x16x32_bf16(false, a, false, b, (short)0, c,
                                                 false, false);
}

// ---------------------------------------------------------------------------
// fp32 -> bf16 conversion (8 elements / thread, b128 in, b128 out)
// ---------------------------------------------------------------------------
__global__ void k_f32_to_bf16(const float* __restrict__ in,
                              bf16_t* __restrict__ out, int n) {
  int i = (blockIdx.x * blockDim.x + threadIdx.x) * 8;
  if (i >= n) return;
  float4 a = *(const float4*)(in + i);
  float4 b = *(const float4*)(in + i + 4);
  v8bf r;
  r[0] = f2bf(a.x); r[1] = f2bf(a.y); r[2] = f2bf(a.z); r[3] = f2bf(a.w);
  r[4] = f2bf(b.x); r[5] = f2bf(b.y); r[6] = f2bf(b.z); r[7] = f2bf(b.w);
  *(v8bf*)(out + i) = r;
}

// ---------------------------------------------------------------------------
// Y[R,E] = X[R,E] @ W[E,E]^T  (bf16 in, bf16 out, optional transposed copy).
// 8 waves: 2 (row) x 4 (col); wave tile 32x64; WG tile 64x256.
// A-frag: rows of X along K (contiguous). B-frag: B[k][n]=W[n][k] -> rows of
// W along K (contiguous). All feeds are global_load_b128.
// ---------------------------------------------------------------------------
__global__ void __launch_bounds__(256)
k_gemm_xwT(const bf16_t* __restrict__ X, const bf16_t* __restrict__ W, int R,
           bf16_t* __restrict__ Y, bf16_t* __restrict__ YT) {
  const int lane = threadIdx.x & 31;
  const int wid  = threadIdx.x >> 5;
  const int m16  = lane & 15;
  const int kh   = lane >> 4;          // 0: lanes 0-15, 1: lanes 16-31
  const int wr = wid >> 2, wc = wid & 3;
  const int rowA = blockIdx.x * 64 + wr * 32;
  const int colB = blockIdx.y * 256 + wc * 64;

  v8f acc[2][4];
#pragma unroll
  for (int a = 0; a < 2; ++a)
#pragma unroll
    for (int b = 0; b < 4; ++b) acc[a][b] = {};

  for (int kb = 0; kb < kE; kb += 32) {
    BV aF[2];
#pragma unroll
    for (int mf = 0; mf < 2; ++mf) {
      const bf16_t* p = X + (size_t)(rowA + mf * 16 + m16) * kE + kb + kh * 8;
      aF[mf].h[0] = *(const v8bf*)(p);
      aF[mf].h[1] = *(const v8bf*)(p + 16);
      __builtin_prefetch(p + 32, 0, 1);   // next K tile -> global_prefetch_b8
    }
#pragma unroll
    for (int nf = 0; nf < 4; ++nf) {
      BV bF;
      const bf16_t* p = W + (size_t)(colB + nf * 16 + m16) * kE + kb + kh * 8;
      bF.h[0] = *(const v8bf*)(p);
      bF.h[1] = *(const v8bf*)(p + 16);
#pragma unroll
      for (int mf = 0; mf < 2; ++mf)
        acc[mf][nf] = wmma_bf16(aF[mf].v, bF.v, acc[mf][nf]);
    }
  }

#pragma unroll
  for (int mf = 0; mf < 2; ++mf)
#pragma unroll
    for (int nf = 0; nf < 4; ++nf)
#pragma unroll
      for (int r = 0; r < 8; ++r) {
        int grow = rowA + mf * 16 + r + 8 * kh;       // C layout: M = r + 8*half
        int gcol = colB + nf * 16 + m16;              // N = lane%16
        bf16_t v = f2bf(acc[mf][nf][r]);
        if (Y)  Y[(size_t)grow * kE + gcol] = v;
        if (YT) YT[(size_t)gcol * R + grow] = v;      // transposed copy for P@V
      }
}

// ---------------------------------------------------------------------------
// Diagonal self score: s[i] = q_i . k_i  (b128 vector loads)
// ---------------------------------------------------------------------------
__global__ void k_selfdot(const bf16_t* __restrict__ q,
                          const bf16_t* __restrict__ kk,
                          float* __restrict__ s) {
  int i = blockIdx.x * blockDim.x + threadIdx.x;
  if (i >= kN) return;
  const bf16_t* qr = q + (size_t)i * kE;
  const bf16_t* kr = kk + (size_t)i * kE;
  float acc = 0.f;
  for (int t = 0; t < kE; t += 8) {
    v8bf qv = *(const v8bf*)(qr + t);
    v8bf kv = *(const v8bf*)(kr + t);
#pragma unroll
    for (int u = 0; u < 8; ++u) acc += bf2f(qv[u]) * bf2f(kv[u]);
  }
  s[i] = acc;
}

// ---------------------------------------------------------------------------
// Fused flash attention + residual + LayerNorm.
// One WG = 8 waves owns 32 Q rows (2 M-frags per wave amortize every K/V
// B-fragment over 32 output rows -> half the L2 traffic of a 16-row WG).
// Wave w: K-slice [64w,64w+64) of E for the S GEMM (partials reduced via
// LDS), E-slice [64w,64w+64) of the O accumulator.
// Self term = initial softmax state. Softmax + LN stats are fully parallel
// (width-8 __shfl_xor reductions, lockstep within a wave).
// ---------------------------------------------------------------------------
__global__ void __launch_bounds__(256)
k_attn(const bf16_t* __restrict__ Q, const bf16_t* __restrict__ K,
       const bf16_t* __restrict__ VT,   // obs_value^T  [E][M]
       const bf16_t* __restrict__ VV,   // v_value      [N][E]
       const float* __restrict__ v_code, const float* __restrict__ gamma,
       const float* __restrict__ beta, const float* __restrict__ sSelf,
       float* __restrict__ out) {
  __shared__ float sPart[8][32][64];               // 64KB, reused as xb[32][512]
  __shared__ __align__(16) bf16_t pTile[32][64];   // softmax tile (bf16 A feed)
  __shared__ float rowM[32], rowL[32], rowA[32], muA[32], rsA[32];

  const int tid  = threadIdx.x;
  const int lane = tid & 31;
  const int wid  = tid >> 5;
  const int m16  = lane & 15;
  const int kh   = lane >> 4;
  const int row0 = blockIdx.x * 32;
  const int e0   = wid * 64;          // O e-slice
  const int kw   = wid * 64;          // S k-slice

  // Preload this wave's Q A-fragments (32 rows x 64 K) once.
  BV qa[2][2];                        // [mf][t]
#pragma unroll
  for (int mf = 0; mf < 2; ++mf)
#pragma unroll
    for (int t = 0; t < 2; ++t) {
      const bf16_t* p =
          Q + (size_t)(row0 + mf * 16 + m16) * kE + kw + t * 32 + kh * 8;
      qa[mf][t].h[0] = *(const v8bf*)(p);
      qa[mf][t].h[1] = *(const v8bf*)(p + 16);
    }

  // O init = v_value (self-term contribution, weight exp(0)=1).
  v8f o[2][4];
#pragma unroll
  for (int mf = 0; mf < 2; ++mf)
#pragma unroll
    for (int nf = 0; nf < 4; ++nf)
#pragma unroll
      for (int r = 0; r < 8; ++r)
        o[mf][nf][r] = bf2f(VV[(size_t)(row0 + mf * 16 + r + 8 * kh) * kE +
                               e0 + nf * 16 + m16]);

  if (tid < 32) {
    rowM[tid] = sSelf[row0 + tid] * kInvT;  // m0 = self score / T
    rowL[tid] = 1.0f;                       // l0 = exp(0)
  }
  __syncthreads();

  for (int jt = 0; jt < kM / 64; ++jt) {
    const int j0 = jt * 64;

    // ---- partial S = Q_kslice @ K_kslice^T -------------------------------
    v8f cs[2][4];
#pragma unroll
    for (int mf = 0; mf < 2; ++mf)
#pragma unroll
      for (int jf = 0; jf < 4; ++jf) cs[mf][jf] = {};
#pragma unroll
    for (int jf = 0; jf < 4; ++jf) {
      const bf16_t* kp = K + (size_t)(j0 + jf * 16 + m16) * kE + kw + kh * 8;
      __builtin_prefetch(kp + (size_t)64 * kE, 0, 1);   // next j-tile
#pragma unroll
      for (int t = 0; t < 2; ++t) {
        BV bF;
        bF.h[0] = *(const v8bf*)(kp + t * 32);
        bF.h[1] = *(const v8bf*)(kp + t * 32 + 16);
#pragma unroll
        for (int mf = 0; mf < 2; ++mf)
          cs[mf][jf] = wmma_bf16(qa[mf][t].v, bF.v, cs[mf][jf]);
      }
    }
#pragma unroll
    for (int mf = 0; mf < 2; ++mf)
#pragma unroll
      for (int jf = 0; jf < 4; ++jf)
#pragma unroll
        for (int r = 0; r < 8; ++r)
          sPart[wid][mf * 16 + r + 8 * kh][jf * 16 + m16] = cs[mf][jf][r];
    __syncthreads();

    // ---- fused cross-wave reduction + online softmax (all 256 threads) ---
    {
      const int m = tid >> 3;            // row 0..31
      const int c = tid & 7;             // 8 threads per row, 8 cols each
      float sv[8];
      float lmax = -3.0e38f;
#pragma unroll
      for (int jj = 0; jj < 8; ++jj) {
        int j = c * 8 + jj;
        float s = 0.f;
#pragma unroll
        for (int w = 0; w < 8; ++w) s += sPart[w][m][j];
        s *= kInvT;
        sv[jj] = s;
        lmax = fmaxf(lmax, s);
      }
#pragma unroll
      for (int off = 1; off < 8; off <<= 1)
        lmax = fmaxf(lmax, __shfl_xor(lmax, off, 8));
      float mOld = rowM[m];
      float mNew = fmaxf(mOld, lmax);
      float a = __expf(mOld - mNew);
      float ps = 0.f;
#pragma unroll
      for (int jj = 0; jj < 8; ++jj) {
        float pv = __expf(sv[jj] - mNew);
        pTile[m][c * 8 + jj] = f2bf(pv);
        ps += pv;
      }
#pragma unroll
      for (int off = 1; off < 8; off <<= 1) ps += __shfl_xor(ps, off, 8);
      if (c == 0) {                       // lockstep within wave: reads above
        rowM[m] = mNew;                   // happen before this write
        rowL[m] = rowL[m] * a + ps;
        rowA[m] = a;
      }
    }
    __syncthreads();

    // ---- rescale O, accumulate P @ V (ds_load_b128 A, global b128 B) -----
    float ar[2][8];
#pragma unroll
    for (int mf = 0; mf < 2; ++mf)
#pragma unroll
      for (int r = 0; r < 8; ++r) ar[mf][r] = rowA[mf * 16 + r + 8 * kh];
#pragma unroll
    for (int mf = 0; mf < 2; ++mf)
#pragma unroll
      for (int nf = 0; nf < 4; ++nf)
#pragma unroll
        for (int r = 0; r < 8; ++r) o[mf][nf][r] *= ar[mf][r];

    BV pA[2][2];
#pragma unroll
    for (int mf = 0; mf < 2; ++mf)
#pragma unroll
      for (int t = 0; t < 2; ++t) {
        pA[mf][t].h[0] = *(const v8bf*)&pTile[mf * 16 + m16][t * 32 + kh * 8];
        pA[mf][t].h[1] =
            *(const v8bf*)&pTile[mf * 16 + m16][t * 32 + 16 + kh * 8];
      }
#pragma unroll
    for (int nf = 0; nf < 4; ++nf) {
      const bf16_t* vp =
          VT + (size_t)(e0 + nf * 16 + m16) * kM + j0 + kh * 8;
      __builtin_prefetch(vp + 64, 0, 1);                // next j-tile
#pragma unroll
      for (int t = 0; t < 2; ++t) {
        BV bF;
        bF.h[0] = *(const v8bf*)(vp + t * 32);
        bF.h[1] = *(const v8bf*)(vp + t * 32 + 16);
#pragma unroll
        for (int mf = 0; mf < 2; ++mf)
          o[mf][nf] = wmma_bf16(pA[mf][t].v, bF.v, o[mf][nf]);
      }
    }
    __syncthreads();
  }

  // ---- epilogue: x = O/l + v_code; LayerNorm -----------------------------
  float* xb = &sPart[0][0][0];   // reuse 64KB as [32][512] fp32
  float linv[2][8];
#pragma unroll
  for (int mf = 0; mf < 2; ++mf)
#pragma unroll
    for (int r = 0; r < 8; ++r)
      linv[mf][r] = 1.0f / rowL[mf * 16 + r + 8 * kh];
#pragma unroll
  for (int mf = 0; mf < 2; ++mf)
#pragma unroll
    for (int nf = 0; nf < 4; ++nf)
#pragma unroll
      for (int r = 0; r < 8; ++r) {
        int m = mf * 16 + r + 8 * kh;
        int e = e0 + nf * 16 + m16;
        xb[m * kE + e] =
            o[mf][nf][r] * linv[mf][r] + v_code[(size_t)(row0 + m) * kE + e];
      }
  __syncthreads();
  {
    const int m = tid >> 3;          // 8 threads per row
    const int c = tid & 7;
    float sum = 0.f;
    for (int i = 0; i < 64; ++i) sum += xb[m * kE + c + 8 * i];
#pragma unroll
    for (int off = 1; off < 8; off <<= 1) sum += __shfl_xor(sum, off, 8);
    float mu = sum * (1.0f / kE);
    float var = 0.f;
    for (int i = 0; i < 64; ++i) {
      float d = xb[m * kE + c + 8 * i] - mu;
      var += d * d;
    }
#pragma unroll
    for (int off = 1; off < 8; off <<= 1) var += __shfl_xor(var, off, 8);
    if (c == 0) {
      muA[m] = mu;
      rsA[m] = rsqrtf(var * (1.0f / kE) + kEps);
    }
  }
  __syncthreads();
#pragma unroll
  for (int ii = 0; ii < 64; ++ii) {
    int idx = tid + 256 * ii;                    // coalesced
    int m = idx >> 9, e = idx & 511;
    out[(size_t)(row0 + m) * kE + e] =
        (xb[idx] - muA[m]) * rsA[m] * gamma[e] + beta[e];
  }
}

// ---------------------------------------------------------------------------
// Host launch
// ---------------------------------------------------------------------------
extern "C" void kernel_launch(void* const* d_in, const int* in_sizes, int n_in,
                              void* d_out, int out_size, void* d_ws, size_t ws_size,
                              hipStream_t stream) {
  (void)in_sizes; (void)n_in; (void)out_size; (void)ws_size;
  const float* v_code   = (const float*)d_in[0];
  const float* obs_code = (const float*)d_in[1];
  const float* Wq       = (const float*)d_in[2];
  const float* Wk       = (const float*)d_in[3];
  const float* Wv       = (const float*)d_in[4];
  const float* gamma    = (const float*)d_in[5];
  const float* beta     = (const float*)d_in[6];
  float* out = (float*)d_out;

  char* p = (char*)d_ws;
  auto carve = [&](size_t bytes) -> char* {
    char* r = p;
    p += (bytes + 255) & ~(size_t)255;
    return r;
  };
  bf16_t* vcb  = (bf16_t*)carve((size_t)kN * kE * 2);  // v_code bf16
  bf16_t* ocb  = (bf16_t*)carve((size_t)kM * kE * 2);  // obs_code bf16
  bf16_t* wqb  = (bf16_t*)carve((size_t)kE * kE * 2);
  bf16_t* wkb  = (bf16_t*)carve((size_t)kE * kE * 2);
  bf16_t* wvb  = (bf16_t*)carve((size_t)kE * kE * 2);
  bf16_t* qb   = (bf16_t*)carve((size_t)kN * kE * 2);  // v_query
  bf16_t* kb   = (bf16_t*)carve((size_t)kN * kE * 2);  // v_key
  bf16_t* vvb  = (bf16_t*)carve((size_t)kN * kE * 2);  // v_value
  bf16_t* okb  = (bf16_t*)carve((size_t)kM * kE * 2);  // obs_key
  bf16_t* ovT  = (bf16_t*)carve((size_t)kM * kE * 2);  // obs_value^T [E][M]
  float*  sSelf = (float*)carve((size_t)kN * 4);

  const int T = 256;
  k_f32_to_bf16<<<(kN * kE) / (8 * T), T, 0, stream>>>(v_code, vcb, kN * kE);
  k_f32_to_bf16<<<(kM * kE) / (8 * T), T, 0, stream>>>(obs_code, ocb, kM * kE);
  k_f32_to_bf16<<<(kE * kE) / (8 * T), T, 0, stream>>>(Wq, wqb, kE * kE);
  k_f32_to_bf16<<<(kE * kE) / (8 * T), T, 0, stream>>>(Wk, wkb, kE * kE);
  k_f32_to_bf16<<<(kE * kE) / (8 * T), T, 0, stream>>>(Wv, wvb, kE * kE);

  dim3 gN(kN / 64, kE / 256);
  dim3 gM(kM / 64, kE / 256);
  k_gemm_xwT<<<gN, T, 0, stream>>>(vcb, wqb, kN, qb, nullptr);
  k_gemm_xwT<<<gN, T, 0, stream>>>(vcb, wkb, kN, kb, nullptr);
  k_gemm_xwT<<<gN, T, 0, stream>>>(vcb, wvb, kN, vvb, nullptr);
  k_gemm_xwT<<<gM, T, 0, stream>>>(ocb, wkb, kM, okb, nullptr);
  k_gemm_xwT<<<gM, T, 0, stream>>>(ocb, wvb, kM, nullptr, ovT);

  k_selfdot<<<kN / T, T, 0, stream>>>(qb, kb, sSelf);

  k_attn<<<kN / 32, T, 0, stream>>>(qb, okb, ovT, vvb, v_code, gamma, beta,
                                    sSelf, out);
}
